// DeepESN_buffer_54580444398289
// MI455X (gfx1250) — matrix operand order repaired and checked by reading, not compiled, not run on hardware
//
#include <hip/hip_runtime.h>
#include <math.h>

// ---------------------------------------------------------------------------
// DeepESN fused kernel for MI455X (gfx1250, wave32, WMMA bf16 -> f32)
//
//  - prep: transpose+convert all weight matrices to column-major bf16 in d_ws
//  - main: persistent recurrence kernel; each workgroup (512 thr = 16 waves)
//          owns 16 batch rows; each wave owns 32 output columns. h0/h1/h2 are
//          carried as f32 in registers (48 VGPRs) + bf16 mirrors in LDS across
//          all T=512 steps. All GEMMs via v_wmma_f32_16x16x32_bf16.
//          A loop-variant zero offset (inline-asm laundered) stops LICM from
//          hoisting the loop-invariant weight loads (which caused scratch
//          spills) while keeping global (not flat) address-space codegen.
//  - readout: tiny [256,1536]x[1536,10] dense readout.
// ---------------------------------------------------------------------------

typedef __attribute__((ext_vector_type(16))) __bf16 v16bf;
typedef __attribute__((ext_vector_type(8)))  __bf16 v8bf;
typedef __attribute__((ext_vector_type(8)))  float  v8f;

#define B_DIM   256
#define T_DIM   512
#define D_DIM   64
#define U_DIM   512
#define NCLS    10
#define HSTR    520   // LDS row stride (elements) for h tiles: 1040B, bank-friendly
#define XSTR    72    // LDS row stride for x tile
#define BLOCK   512   // 16 wave32s
#define JT      2     // 16x16 column tiles per wave (32 cols)

// bf16-element offsets of the transposed weights inside d_ws
#define OFF_IN0 0
#define OFF_W0  32768            // 512*64
#define OFF_IN1 294912           // + 512*512
#define OFF_W1  557056
#define OFF_IN2 819200
#define OFF_W2  1081344
#define WT_TOTAL_BYTES 2686976   // (1081344 + 262144) * 2

// Loop-variant zero: blocks LICM on weight loads without destroying the
// global address-space provenance of the weight base pointer.
__device__ __forceinline__ size_t launder_zero() {
  size_t z = 0;
  asm volatile("" : "+s"(z));
  return z;
}

// ---------------------------------------------------------------------------
// WMMA helpers
// ---------------------------------------------------------------------------
__device__ __forceinline__ v8f wmma_bf16(v16bf a, v16bf b, v8f c) {
  return __builtin_amdgcn_wmma_f32_16x16x32_bf16(
      /*neg_a=*/false, a, /*neg_b=*/false, b,
      /*c_mod=*/(short)0, c, /*reuse_a=*/false, /*reuse_b=*/false);
}

// A-matrix 16x32 bf16 fragment from an LDS tile buf[16][stride].
// lanes 0-15: M=lane, K=0..7 / 16..23; lanes 16-31: M=lane-16, K=8..15 / 24..31.
__device__ __forceinline__ v16bf load_a_frag(const __bf16* buf, int stride,
                                             int kc, int lane) {
  const int m    = lane & 15;
  const int koff = kc * 32 + ((lane & 16) ? 8 : 0);
  const __bf16* p = buf + m * stride + koff;
  v16bf a;
  ((v8bf*)&a)[0] = *(const v8bf*)(p);        // 16B ds_load
  ((v8bf*)&a)[1] = *(const v8bf*)(p + 16);   // 16B ds_load
  return a;
}

// B-matrix 32x16 bf16 fragment from column-major weights wT[n][k] (row len K).
// lanes 0-15: N=lane, K=0..7 / 16..23; lanes 16-31: N=lane-16, K=8..15 / 24..31.
__device__ __forceinline__ v16bf load_b_frag(const __bf16* __restrict__ wT,
                                             int krow, int ncol, int kc, int lane) {
  const int n    = ncol + (lane & 15);
  const int koff = kc * 32 + ((lane & 16) ? 8 : 0);
  const __bf16* p = wT + (size_t)n * krow + koff;
  v16bf b;
  ((v8bf*)&b)[0] = *(const v8bf*)(p);        // 16B global load (L2-resident)
  ((v8bf*)&b)[1] = *(const v8bf*)(p + 16);
  return b;
}

// One ESN layer step for this wave's 16x32 output slab:
//   acc = bias + drvA @ drvWT + recA @ recWT ; h = 0.1*h + 0.9*tanh(acc)
__device__ __forceinline__ void layer_step(
    float hreg_l[JT][8], const float bias_l[JT],
    const __bf16* drvA, int drvStride, int drvKC,
    const __bf16* __restrict__ drvWT, int drvK,
    const __bf16* recA, const __bf16* __restrict__ recWT,
    __bf16 (*hout)[HSTR], int n0, int lane)
{
  v8f acc[JT];
#pragma unroll
  for (int j = 0; j < JT; ++j) {
#pragma unroll
    for (int r = 0; r < 8; ++r) acc[j][r] = bias_l[j];
  }

  // input-drive GEMM
  for (int kc = 0; kc < drvKC; ++kc) {
    v16bf a = load_a_frag(drvA, drvStride, kc, lane);
#pragma unroll
    for (int j = 0; j < JT; ++j) {
      v16bf b = load_b_frag(drvWT, drvK, n0 + 16 * j, kc, lane);
      acc[j] = wmma_bf16(a, b, acc[j]);
    }
  }

  // recurrent GEMM, K = 512 in 16 chunks of 32
  for (int kc = 0; kc < U_DIM / 32; ++kc) {
    v16bf a = load_a_frag(recA, HSTR, kc, lane);
#pragma unroll
    for (int j = 0; j < JT; ++j) {
      v16bf b = load_b_frag(recWT, U_DIM, n0 + 16 * j, kc, lane);
      acc[j] = wmma_bf16(a, b, acc[j]);
    }
  }

  // leaky tanh update; f32 master in registers, bf16 mirror to LDS
  const int mofs = (lane & 16) ? 8 : 0;
  const int ncol = lane & 15;
#pragma unroll
  for (int j = 0; j < JT; ++j) {
#pragma unroll
    for (int r = 0; r < 8; ++r) {
      float v = hreg_l[j][r] * 0.1f + 0.9f * tanhf(acc[j][r]);
      hreg_l[j][r] = v;
      hout[r + mofs][n0 + 16 * j + ncol] = (__bf16)v;
    }
  }
}

// ---------------------------------------------------------------------------
// Main persistent recurrence kernel: grid = 16 WGs x 512 threads (16 waves).
// WG b owns batch rows [16b, 16b+16); wave w owns output cols [32w, 32w+32).
// ---------------------------------------------------------------------------
__launch_bounds__(BLOCK, 1)
__global__ void deep_esn_recur(const float* __restrict__ x,
                               const __bf16* __restrict__ wt,
                               const float* __restrict__ b0,
                               const float* __restrict__ b1,
                               const float* __restrict__ b2,
                               float* __restrict__ states)
{
  __shared__ __align__(16) __bf16 hbuf[3][2][16][HSTR];  // ~100 KB of 320 KB
  __shared__ __align__(16) __bf16 xbuf[16][XSTR];

  const int tid   = threadIdx.x;
  const int lane  = tid & 31;
  const int wave  = tid >> 5;       // 0..15
  const int n0    = wave * (16 * JT);
  const int bbase = blockIdx.x * 16;

  // per-(layer, col-tile) bias, broadcast over rows of the C fragment
  float biasv[3][JT];
#pragma unroll
  for (int j = 0; j < JT; ++j) {
    const int n = n0 + 16 * j + (lane & 15);
    biasv[0][j] = b0[n];
    biasv[1][j] = b1[n];
    biasv[2][j] = b2[n];
  }

  // f32 master copies of h0/h1/h2 fragments (16 rows x 32 cols per wave)
  float hreg[3][JT][8];
#pragma unroll
  for (int l = 0; l < 3; ++l)
#pragma unroll
    for (int j = 0; j < JT; ++j)
#pragma unroll
      for (int r = 0; r < 8; ++r) hreg[l][j][r] = 0.0f;

  // zero the parity-0 LDS h buffers (h(t=0) = 0)
  for (int i = tid; i < 16 * HSTR; i += BLOCK) {
    const int m = i / HSTR, k = i % HSTR;
    hbuf[0][0][m][k] = (__bf16)0.0f;
    hbuf[1][0][m][k] = (__bf16)0.0f;
    hbuf[2][0][m][k] = (__bf16)0.0f;
  }
  __syncthreads();

  int p = 0;
  for (int t = 0; t < T_DIM; ++t) {
    // loop-variant zero offset: weight loads stay in-loop (L2-resident
    // global_load_b128) instead of being hoisted+spilled to scratch
    const size_t z = launder_zero();
    const __bf16* WT_in0 = wt + OFF_IN0 + z;   // [512][64]
    const __bf16* WT_w0  = wt + OFF_W0  + z;   // [512][512]
    const __bf16* WT_in1 = wt + OFF_IN1 + z;
    const __bf16* WT_w1  = wt + OFF_W1  + z;
    const __bf16* WT_in2 = wt + OFF_IN2 + z;
    const __bf16* WT_w2  = wt + OFF_W2  + z;

    // stage x_t tile [16 x 64] f32 -> bf16 into LDS
    for (int i = tid; i < 16 * D_DIM; i += BLOCK) {
      const int m = i >> 6, d = i & 63;
      xbuf[m][d] = (__bf16)x[((size_t)(bbase + m) * T_DIM + t) * D_DIM + d];
    }
    __syncthreads();

    // layer 0: drive from x_t (K=64), recur from h0[p]
    layer_step(hreg[0], biasv[0],
               &xbuf[0][0], XSTR, D_DIM / 32, WT_in0, D_DIM,
               &hbuf[0][p][0][0], WT_w0,
               hbuf[0][p ^ 1], n0, lane);
    __syncthreads();

    // layer 1: drive from h0_new (K=512), recur from h1[p]
    layer_step(hreg[1], biasv[1],
               &hbuf[0][p ^ 1][0][0], HSTR, U_DIM / 32, WT_in1, U_DIM,
               &hbuf[1][p][0][0], WT_w1,
               hbuf[1][p ^ 1], n0, lane);
    __syncthreads();

    // layer 2: drive from h1_new (K=512), recur from h2[p]
    layer_step(hreg[2], biasv[2],
               &hbuf[1][p ^ 1][0][0], HSTR, U_DIM / 32, WT_in2, U_DIM,
               &hbuf[2][p][0][0], WT_w2,
               hbuf[2][p ^ 1], n0, lane);
    __syncthreads();

    p ^= 1;
  }

  // write final states [B, 3U] in f32 for the readout
  const int mofs = (lane & 16) ? 8 : 0;
  const int ncol = lane & 15;
#pragma unroll
  for (int l = 0; l < 3; ++l) {
#pragma unroll
    for (int j = 0; j < JT; ++j) {
#pragma unroll
      for (int r = 0; r < 8; ++r) {
        const int m = r + mofs;
        const int n = n0 + 16 * j + ncol;
        states[(size_t)(bbase + m) * (3 * U_DIM) + l * U_DIM + n] = hreg[l][j][r];
      }
    }
  }
}

// ---------------------------------------------------------------------------
// Prep: W[k][n] (f32, row-major) -> WT[n][k] (bf16, column-major rows)
// ---------------------------------------------------------------------------
__global__ void conv_transpose(const float* __restrict__ src,
                               __bf16* __restrict__ dst, int K, int N)
{
  const int i = blockIdx.x * blockDim.x + threadIdx.x;
  if (i >= K * N) return;
  const int n = i / K;
  const int k = i % K;
  dst[(size_t)n * K + k] = (__bf16)src[(size_t)k * N + n];
}

// ---------------------------------------------------------------------------
// Readout: scores[B,10] = states[B,1536] @ Wout[1536,10] + bout (tiny)
// ---------------------------------------------------------------------------
__global__ void readout(const float* __restrict__ states,
                        const float* __restrict__ Wout,
                        const float* __restrict__ bout,
                        float* __restrict__ out)
{
  const int i = blockIdx.x * blockDim.x + threadIdx.x;
  if (i >= B_DIM * NCLS) return;
  const int b = i / NCLS;
  const int c = i % NCLS;
  float s = bout[c];
  const float* row = states + (size_t)b * (3 * U_DIM);
  for (int k = 0; k < 3 * U_DIM; ++k) s += row[k] * Wout[(size_t)k * NCLS + c];
  out[i] = s;
}

// ---------------------------------------------------------------------------
extern "C" void kernel_launch(void* const* d_in, const int* in_sizes, int n_in,
                              void* d_out, int out_size, void* d_ws, size_t ws_size,
                              hipStream_t stream)
{
  const float* x    = (const float*)d_in[0];
  const float* Win0 = (const float*)d_in[1];
  const float* W0   = (const float*)d_in[2];
  const float* b0   = (const float*)d_in[3];
  const float* Win1 = (const float*)d_in[4];
  const float* W1   = (const float*)d_in[5];
  const float* b1   = (const float*)d_in[6];
  const float* Win2 = (const float*)d_in[7];
  const float* W2   = (const float*)d_in[8];
  const float* b2   = (const float*)d_in[9];
  const float* Wout = (const float*)d_in[10];
  const float* bout = (const float*)d_in[11];

  __bf16* wt     = (__bf16*)d_ws;
  float*  states = (float*)((char*)d_ws + WT_TOTAL_BYTES);

  // transpose + bf16-convert all weights (cheap, once per call)
  conv_transpose<<<(D_DIM * U_DIM + 255) / 256, 256, 0, stream>>>(Win0, wt + OFF_IN0, D_DIM, U_DIM);
  conv_transpose<<<(U_DIM * U_DIM + 255) / 256, 256, 0, stream>>>(W0,   wt + OFF_W0,  U_DIM, U_DIM);
  conv_transpose<<<(U_DIM * U_DIM + 255) / 256, 256, 0, stream>>>(Win1, wt + OFF_IN1, U_DIM, U_DIM);
  conv_transpose<<<(U_DIM * U_DIM + 255) / 256, 256, 0, stream>>>(W1,   wt + OFF_W1,  U_DIM, U_DIM);
  conv_transpose<<<(U_DIM * U_DIM + 255) / 256, 256, 0, stream>>>(Win2, wt + OFF_IN2, U_DIM, U_DIM);
  conv_transpose<<<(U_DIM * U_DIM + 255) / 256, 256, 0, stream>>>(W2,   wt + OFF_W2,  U_DIM, U_DIM);

  // fused 3-layer recurrence over all T=512 steps
  deep_esn_recur<<<B_DIM / 16, BLOCK, 0, stream>>>(x, wt, b0, b1, b2, states);

  // ridge readout
  readout<<<(B_DIM * NCLS + 255) / 256, 256, 0, stream>>>(states, Wout, bout, (float*)d_out);
}
